// Nerad_48009144434941
// MI455X (gfx1250) — compile-verified, matrix operand order checked by reading, not run on hardware
//
#include <hip/hip_runtime.h>

// ---------------------------------------------------------------------------
// Nerad fused MLP (12 -> 256 -> 256 -> 256 -> 3) via CDNA5 WMMA f16 (wave32).
// Weights are pre-converted once to f16 in B-fragment-native transposed layout
// (Wt[n][k], K contiguous) so the GEMM loop is pure {global_load_b128,
// ds_load_b128, v_wmma} with immediate offsets -- no in-loop transpose/convert.
// ---------------------------------------------------------------------------

typedef _Float16 half8 __attribute__((ext_vector_type(8)));
typedef _Float16 v16h  __attribute__((ext_vector_type(16)));
typedef float    v8f   __attribute__((ext_vector_type(8)));

#define HID    256
#define ROWS   64      // batch rows per block
#define KSTEP  32      // WMMA K per step (f16)
#define NTHR   256     // 8 wave32 waves
#define IN_DIM 12

union AFrag { v16h v; half8 h[2]; };

// ---------------------------------------------------------------------------
// One-time weight conversion: fp32 [K][N] -> f16 transposed [N][K] (padded).
// Tiny (143K elements); runs once per launch, keeps weights L2-resident in
// exactly the layout the WMMA B-fragment wants (two contiguous b128 per frag).
// ---------------------------------------------------------------------------
__global__ void __launch_bounds__(256)
nerad_prep_weights(const float* __restrict__ W0, const float* __restrict__ W1,
                   const float* __restrict__ W2, const float* __restrict__ W3,
                   _Float16* __restrict__ W0t, _Float16* __restrict__ W1t,
                   _Float16* __restrict__ W2t, _Float16* __restrict__ W3t)
{
    int i = blockIdx.x * 256 + threadIdx.x;
    if (i < HID * KSTEP) {                       // W0t [256 n][32 k], K pad 12->32
        int n = i >> 5, k = i & 31;
        W0t[i] = (_Float16)((k < IN_DIM) ? W0[k * HID + n] : 0.0f);
    }
    if (i < HID * HID) {                         // W1t/W2t [256 n][256 k]
        int n = i >> 8, k = i & 255;
        W1t[i] = (_Float16)W1[k * HID + n];
        W2t[i] = (_Float16)W2[k * HID + n];
    }
    if (i < 16 * HID) {                          // W3t [16 n][256 k], N pad 3->16
        int n = i >> 8, k = i & 255;
        W3t[i] = (_Float16)((n < 3) ? W3[k * 3 + n] : 0.0f);
    }
}

// ---------------------------------------------------------------------------
// One 256->256 hidden layer. B-fragments stream straight from global (L2-hot,
// immediate offsets off one per-lane base); A-fragments from LDS activations.
// No per-k-step barriers: only the in-place activation update synchronizes.
// ---------------------------------------------------------------------------
__device__ void dense_hidden(_Float16* act, const _Float16* __restrict__ Wt,
                             const float* __restrict__ b,
                             int lane, int rg, int cg)
{
    v8f acc[2][4] = {};
    const int klw = (lane < 16) ? 0 : 8;   // K sub-chunk select (A and B alike)
    const _Float16* wbase = Wt  + (cg * 64 + (lane & 15)) * HID + klw;
    const _Float16* abase = act + (rg * 32 + (lane & 15)) * HID + klw;

    #pragma unroll 4
    for (int kb = 0; kb < HID; kb += KSTEP) {
        AFrag a[2];
        #pragma unroll
        for (int mt = 0; mt < 2; ++mt) {
            a[mt].h[0] = *(const half8*)(abase + mt * 16 * HID + kb);
            a[mt].h[1] = *(const half8*)(abase + mt * 16 * HID + kb + 16);
        }
        #pragma unroll
        for (int nt = 0; nt < 4; ++nt) {
            AFrag bf;
            bf.h[0] = *(const half8*)(wbase + nt * 16 * HID + kb);
            bf.h[1] = *(const half8*)(wbase + nt * 16 * HID + kb + 16);
            #pragma unroll
            for (int mt = 0; mt < 2; ++mt)
                acc[mt][nt] = __builtin_amdgcn_wmma_f32_16x16x32_f16(
                    false, a[mt].v, false, bf.v, (short)0, acc[mt][nt],
                    false, false);
        }
    }
    __syncthreads();                     // all waves done reading act
    #pragma unroll
    for (int mt = 0; mt < 2; ++mt)
        #pragma unroll
        for (int nt = 0; nt < 4; ++nt) {
            int n = cg * 64 + nt * 16 + (lane & 15);
            float bias = b[n];
            int mb = rg * 32 + mt * 16 + ((lane < 16) ? 0 : 8);
            #pragma unroll
            for (int r = 0; r < 8; ++r) {
                float v = acc[mt][nt][r] + bias;
                act[(mb + r) * HID + n] = (_Float16)(v > 0.0f ? v : 0.0f);
            }
        }
    __syncthreads();
}

__global__ void __launch_bounds__(NTHR)
nerad_mlp_kernel(const float* __restrict__ pts, const float* __restrict__ dirs,
                 const float* __restrict__ nrm, const float* __restrict__ alb,
                 const _Float16* __restrict__ W0t, const float* __restrict__ b0,
                 const _Float16* __restrict__ W1t, const float* __restrict__ b1,
                 const _Float16* __restrict__ W2t, const float* __restrict__ b2,
                 const _Float16* __restrict__ W3t, const float* __restrict__ b3,
                 float* __restrict__ out_net, int rows_total)
{
    __shared__ __align__(16) _Float16 act[ROWS * HID];   // 32 KB activations

    const int tid  = threadIdx.x;
    const int lane = tid & 31;
    const int wv   = tid >> 5;     // wave 0..7
    const int rg   = wv >> 2;      // row group 0..1 (32 rows each)
    const int cg   = wv & 3;       // col group 0..3 (64 cols each)
    const int row0 = blockIdx.x * ROWS;
    const int klw  = (lane < 16) ? 0 : 8;

    // ---- stage input features: [64 x 12] f32 -> [64 x 32] f16 (zero padded)
    for (int idx = tid; idx < ROWS * KSTEP; idx += NTHR) {
        int r = idx >> 5, f = idx & 31;
        int gr = row0 + r;
        float v = 0.0f;
        if (gr < rows_total && f < IN_DIM) {
            const float* s = (f < 3) ? pts : (f < 6) ? dirs : (f < 9) ? nrm : alb;
            v = s[gr * 3 + (f % 3)];
        }
        act[r * HID + f] = (_Float16)v;
    }
    __syncthreads();

    // ---- layer 0: [64x32] @ [32x256] (single k-step, W0t is [256][32])
    {
        v8f acc[2][4] = {};
        AFrag a[2];
        const _Float16* abase = act + (rg * 32 + (lane & 15)) * HID + klw;
        #pragma unroll
        for (int mt = 0; mt < 2; ++mt) {
            a[mt].h[0] = *(const half8*)(abase + mt * 16 * HID);
            a[mt].h[1] = *(const half8*)(abase + mt * 16 * HID + 16);
        }
        const _Float16* wbase = W0t + (cg * 64 + (lane & 15)) * KSTEP + klw;
        #pragma unroll
        for (int nt = 0; nt < 4; ++nt) {
            AFrag bf;
            bf.h[0] = *(const half8*)(wbase + nt * 16 * KSTEP);
            bf.h[1] = *(const half8*)(wbase + nt * 16 * KSTEP + 16);
            #pragma unroll
            for (int mt = 0; mt < 2; ++mt)
                acc[mt][nt] = __builtin_amdgcn_wmma_f32_16x16x32_f16(
                    false, a[mt].v, false, bf.v, (short)0, acc[mt][nt],
                    false, false);
        }
        __syncthreads();
        #pragma unroll
        for (int mt = 0; mt < 2; ++mt)
            #pragma unroll
            for (int nt = 0; nt < 4; ++nt) {
                int n = cg * 64 + nt * 16 + (lane & 15);
                float bias = b0[n];
                int mb = rg * 32 + mt * 16 + ((lane < 16) ? 0 : 8);
                #pragma unroll
                for (int r = 0; r < 8; ++r) {
                    float v = acc[mt][nt][r] + bias;
                    act[(mb + r) * HID + n] = (_Float16)(v > 0.0f ? v : 0.0f);
                }
            }
        __syncthreads();
    }

    // ---- layers 1 & 2: 256 -> 256, relu
    dense_hidden(act, W1t, b1, lane, rg, cg);
    dense_hidden(act, W2t, b2, lane, rg, cg);

    // ---- head: [64x256] @ [256x16(pad of 3)], softplus; cg==0 waves only
    if (cg == 0) {                               // wave-uniform branch
        v8f acc3[2] = {};
        const _Float16* abase = act + (rg * 32 + (lane & 15)) * HID + klw;
        const _Float16* wbase = W3t + (lane & 15) * HID + klw;
        #pragma unroll
        for (int kb = 0; kb < HID; kb += KSTEP) {
            AFrag bf;
            bf.h[0] = *(const half8*)(wbase + kb);
            bf.h[1] = *(const half8*)(wbase + kb + 16);
            #pragma unroll
            for (int mt = 0; mt < 2; ++mt) {
                AFrag a;
                a.h[0] = *(const half8*)(abase + mt * 16 * HID + kb);
                a.h[1] = *(const half8*)(abase + mt * 16 * HID + kb + 16);
                acc3[mt] = __builtin_amdgcn_wmma_f32_16x16x32_f16(
                    false, a.v, false, bf.v, (short)0, acc3[mt], false, false);
            }
        }
        int n = lane & 15;
        if (n < 3) {
            float bias = b3[n];
            #pragma unroll
            for (int mt = 0; mt < 2; ++mt) {
                int mb = rg * 32 + mt * 16 + ((lane < 16) ? 0 : 8);
                #pragma unroll
                for (int r = 0; r < 8; ++r) {
                    int gr = row0 + mb + r;
                    if (gr < rows_total) {
                        float x = acc3[mt][r] + bias;
                        float sp = (x > 20.0f) ? x : log1pf(expf(x));
                        out_net[gr * 3 + n] = sp;     // softplus radiance
                    }
                }
            }
        }
    }
}

// ---------------------------------------------------------------------------
// Elementwise combine: masking, block-mean over m, validity OR, residual.
// ---------------------------------------------------------------------------
__global__ void __launch_bounds__(256)
nerad_combine_kernel(const float* __restrict__ thr, const float* __restrict__ E,
                     const float* __restrict__ bw,  const float* __restrict__ bsr,
                     const float* __restrict__ esr,
                     const float* __restrict__ net1, const float* __restrict__ net2,
                     const unsigned char* __restrict__ active,
                     const unsigned char* __restrict__ sv1,
                     const unsigned char* __restrict__ sv2,
                     const unsigned char* __restrict__ vri,
                     const unsigned char* __restrict__ nn,
                     float* __restrict__ out, int n, int m)
{
    int i = blockIdx.x * blockDim.x + threadIdx.x;
    if (i >= n) return;

    bool act_i = active[i] != 0;
    bool s1    = sv1[i] != 0;
    bool lhs_m = act_i && s1;

    float t[3] = { thr[i*3+0], thr[i*3+1], thr[i*3+2] };
    float L[3] = { 0.f, 0.f, 0.f };
    if (lhs_m) {
        L[0] = t[0] * net1[i*3+0];
        L[1] = t[1] * net1[i*3+1];
        L[2] = t[2] * net1[i*3+2];
    }

    float R[3] = { 0.f, 0.f, 0.f };
    bool valid = vri[i] != 0;
    for (int j = 0; j < m; ++j) {
        int k = i * m + j;
        bool rv = s1 && (sv2[k] != 0);          // active_next & si_valid2
        #pragma unroll
        for (int c = 0; c < 3; ++c) {
            float rn = rv ? net2[k*3+c] : 0.f;
            R[c] += rn * (t[c] * bw[k*3+c]) + bsr[k*3+c] + esr[k*3+c];
        }
        valid = valid || (act_i && s1 && (nn[k] != 0));
    }
    float inv = 1.0f / (float)m;
    #pragma unroll
    for (int c = 0; c < 3; ++c) R[c] *= inv;

    float alpha = valid ? 1.0f : 0.0f;
    #pragma unroll
    for (int c = 0; c < 3; ++c) {
        float rgb = valid ? (E[i*3+c] + R[c]) : 0.0f;
        float aov = valid ? (E[i*3+c] + L[c]) : 0.0f;
        float d   = L[c] - R[c];
        float res = valid ? d * d : 0.0f;
        out[i*10 + 0 + c] = rgb;
        out[i*10 + 3 + c] = aov;
        out[i*10 + 7 + c] = res;
    }
    out[i*10 + 6] = alpha;
}

// ---------------------------------------------------------------------------
extern "C" void kernel_launch(void* const* d_in, const int* in_sizes, int n_in,
                              void* d_out, int out_size, void* d_ws, size_t ws_size,
                              hipStream_t stream)
{
    const float* throughput = (const float*)d_in[0];
    const float* pts1       = (const float*)d_in[1];
    const float* dirs1      = (const float*)d_in[2];
    const float* normals1   = (const float*)d_in[3];
    const float* albedo1    = (const float*)d_in[4];
    const float* E          = (const float*)d_in[5];
    const float* bsdf_w     = (const float*)d_in[6];
    const float* pts2       = (const float*)d_in[7];
    const float* dirs2      = (const float*)d_in[8];
    const float* normals2   = (const float*)d_in[9];
    const float* albedo2    = (const float*)d_in[10];
    const float* bsdf_res   = (const float*)d_in[11];
    const float* em_res     = (const float*)d_in[12];
    const float* W0 = (const float*)d_in[13]; const float* b0 = (const float*)d_in[14];
    const float* W1 = (const float*)d_in[15]; const float* b1 = (const float*)d_in[16];
    const float* W2 = (const float*)d_in[17]; const float* b2 = (const float*)d_in[18];
    const float* W3 = (const float*)d_in[19]; const float* b3 = (const float*)d_in[20];
    const unsigned char* active = (const unsigned char*)d_in[21];
    const unsigned char* sv1    = (const unsigned char*)d_in[22];
    const unsigned char* sv2    = (const unsigned char*)d_in[23];
    const unsigned char* vri    = (const unsigned char*)d_in[24];
    const unsigned char* nnul   = (const unsigned char*)d_in[25];

    const int N = in_sizes[0] / 3;     // 65536
    const int M = in_sizes[6] / 3;     // 262144
    const int m = M / N;               // 4 (capture-safe; no device read)

    // workspace layout: net1 | net2 | f16 weights (fragment-transposed)
    float* net1 = (float*)d_ws;                       // N*3 f32
    float* net2 = net1 + (size_t)N * 3;               // M*3 f32
    _Float16* W0t = (_Float16*)(net2 + (size_t)M * 3);    // 256*32
    _Float16* W1t = W0t + HID * KSTEP;                    // 256*256
    _Float16* W2t = W1t + HID * HID;                      // 256*256
    _Float16* W3t = W2t + HID * HID;                      // 16*256
    float* outp = (float*)d_out;

    // one-time f16 transposed weight build (L2-resident afterwards)
    nerad_prep_weights<<<(HID * HID + 255) / 256, 256, 0, stream>>>(
        W0, W1, W2, W3, W0t, W1t, W2t, W3t);

    // LHS batch MLP
    nerad_mlp_kernel<<<(N + ROWS - 1) / ROWS, NTHR, 0, stream>>>(
        pts1, dirs1, normals1, albedo1,
        W0t, b0, W1t, b1, W2t, b2, W3t, b3, net1, N);
    // RHS batch MLP
    nerad_mlp_kernel<<<(M + ROWS - 1) / ROWS, NTHR, 0, stream>>>(
        pts2, dirs2, normals2, albedo2,
        W0t, b0, W1t, b1, W2t, b2, W3t, b3, net2, M);
    // combine / reduce / mask
    nerad_combine_kernel<<<(N + 255) / 256, 256, 0, stream>>>(
        throughput, E, bsdf_w, bsdf_res, em_res, net1, net2,
        active, sv1, sv2, vri, nnul, outp, N, m);
}
